// ContrastCELoss_10179072491789
// MI455X (gfx1250) — matrix operand order repaired and checked by reading, not compiled, not run on hardware
//
#include <hip/hip_runtime.h>
#include <hip/hip_bf16.h>

typedef __attribute__((ext_vector_type(16))) _Float16 v16h;
typedef __attribute__((ext_vector_type(8)))  _Float16 v8h;
typedef __attribute__((ext_vector_type(8)))  float    v8f;

#define NPIX      131072      // H*W = 64*2048
#define NCLS      19          // classes 1..19
#define NGROUP    38          // B * NCLS
#define NVIEW     26          // min(1024 // 38, 100)
#define KHARD     13
#define TROWS     988         // NGROUP * NVIEW
#define TPAD      992         // round up to 16
#define DDIM      256
#define INV_TEMP  10.0f       // 1 / 0.1
#define LOSS_SCALE 1.4285714285714286f  // TEMP / BASE_TEMP

__device__ __forceinline__ unsigned int hash_u32(unsigned int x) {
    x ^= x >> 16; x *= 0x7feb352dU;
    x ^= x >> 15; x *= 0x846ca68bU;
    x ^= x >> 16;
    return x;
}

// branchless sorted-descending insert into a 13-entry register list
__device__ __forceinline__ void topk_insert(unsigned long long (&list)[KHARD],
                                            unsigned long long key) {
#pragma unroll
    for (int q = 0; q < KHARD; ++q) {
        unsigned long long cur = list[q];
        if (key > cur) { list[q] = key; key = cur; }
    }
}

// ---------------------------------------------------------------------------
// Kernel 1: per-(batch, class) top-13 hard / top-13 easy pixel selection
// ---------------------------------------------------------------------------
__global__ void ccel_select(const int* __restrict__ labels,
                            const int* __restrict__ predict,
                            int* __restrict__ idx_out) {
    const int g   = blockIdx.x;            // 0..37
    const int b   = g / NCLS;
    const int cls = (g % NCLS) + 1;
    const int t   = threadIdx.x;           // 0..255

    __shared__ unsigned long long smem[256 * KHARD];

    unsigned long long hardL[KHARD], easyL[KHARD];
#pragma unroll
    for (int q = 0; q < KHARD; ++q) { hardL[q] = 0ULL; easyL[q] = 0ULL; }

    const int* lab = labels  + (size_t)b * NPIX;
    const int* prd = predict + (size_t)b * NPIX;

    for (int n = t; n < NPIX; n += 256) {
        int l = lab[n];
        if (l != cls) continue;
        unsigned int h = hash_u32((unsigned int)g * 0x9E3779B9U + (unsigned int)n * 2654435761U);
        unsigned long long key = ((unsigned long long)h << 32) | (unsigned int)n;
        if (prd[n] != cls) topk_insert(hardL, key);   // gt==cls, prediction wrong
        else               topk_insert(easyL, key);   // gt==cls, prediction right
    }

#pragma unroll
    for (int phase = 0; phase < 2; ++phase) {
#pragma unroll
        for (int q = 0; q < KHARD; ++q)
            smem[t * KHARD + q] = (phase == 0) ? hardL[q] : easyL[q];
        __syncthreads();
        for (int off = 128; off >= 1; off >>= 1) {
            if (t < off) {
                unsigned long long out[KHARD];
                int ia = 0, ib = 0;
#pragma unroll
                for (int q = 0; q < KHARD; ++q) {
                    unsigned long long av = smem[t * KHARD + ia];
                    unsigned long long bv = smem[(t + off) * KHARD + ib];
                    if (av >= bv) { out[q] = av; ++ia; } else { out[q] = bv; ++ib; }
                }
#pragma unroll
                for (int q = 0; q < KHARD; ++q) smem[t * KHARD + q] = out[q];
            }
            __syncthreads();
        }
        if (t < KHARD)
            idx_out[g * NVIEW + phase * KHARD + t] = (int)(smem[t] & 0xFFFFFFFFULL);
        __syncthreads();
    }
}

// ---------------------------------------------------------------------------
// Kernel 2: gather selected feature rows, convert f32 -> f16, zero-pad rows
// ---------------------------------------------------------------------------
__global__ void ccel_gather(const float* __restrict__ feats,
                            const int* __restrict__ idx,
                            _Float16* __restrict__ cf) {
    const int r = blockIdx.x;              // 0..991
    const int t = threadIdx.x;             // 0..63 (4 channels each)
    if (r >= TROWS) {
#pragma unroll
        for (int k = 0; k < 4; ++k) cf[(size_t)r * DDIM + t * 4 + k] = (_Float16)0.0f;
        return;
    }
    const int a = r / NVIEW;               // group
    const int v = r % NVIEW;               // view
    const int b = a / NCLS;
    const int n = idx[a * NVIEW + v];
    const float* src = feats + (size_t)b * DDIM * NPIX + n;
#pragma unroll
    for (int k = 0; k < 4; ++k) {
        int d = t * 4 + k;
        cf[(size_t)r * DDIM + d] = (_Float16)src[(size_t)d * NPIX];
    }
}

// ---------------------------------------------------------------------------
// Kernel 3: logits = (cf @ cf^T) / TEMP  via v_wmma_f32_16x16x32_f16
// One wave32 per 16x16 tile; K-loop over D=256 in steps of 32.
// All operand chunks are 16-byte aligned -> explicit b128 vector loads.
// ---------------------------------------------------------------------------
__global__ void __launch_bounds__(32)
ccel_gemm(const _Float16* __restrict__ cf, float* __restrict__ S) {
    const int ib = blockIdx.x * 16;        // tile row base
    const int jb = blockIdx.y * 16;        // tile col base
    const int l  = threadIdx.x;            // 0..31
    const int m  = l & 15;
    const int h  = l >> 4;

    v8f acc = {};
#pragma unroll
    for (int k = 0; k < DDIM; k += 32) {
        // A tile (ISA 16-bit A layout): lane (m,h) holds
        //   halves 0..7  = cf[ib+m][k + 8h .. k + 8h + 7]
        //   halves 8..15 = cf[ib+m][k + 16 + 8h .. k + 16 + 8h + 7]
        const _Float16* pa = cf + (size_t)(ib + m) * DDIM + k + 8 * h;
        v8h a_lo = *(const v8h*)(pa);
        v8h a_hi = *(const v8h*)(pa + 16);
        v16h av = __builtin_shufflevector(a_lo, a_hi,
                                          0, 1, 2, 3, 4, 5, 6, 7,
                                          8, 9, 10, 11, 12, 13, 14, 15);

        // B tile = cf^T[k..k+31, jb..jb+15] (ISA 16-bit B layout): lane (m,h)
        // holds K values k + 8h .. k + 8h + 7 and k + 16 + 8h .. k + 16 + 8h + 7 of column jb+m.
        const _Float16* pb = cf + (size_t)(jb + m) * DDIM + k + 8 * h;
        v8h b_lo = *(const v8h*)(pb);
        v8h b_hi = *(const v8h*)(pb + 16);
        v16h bv = __builtin_shufflevector(b_lo, b_hi,
                                          0, 1, 2, 3, 4, 5, 6, 7,
                                          8, 9, 10, 11, 12, 13, 14, 15);

        acc = __builtin_amdgcn_wmma_f32_16x16x32_f16(
            /*neg_a=*/false, av, /*neg_b=*/false, bv,
            /*c_mod=*/(short)0, acc, /*reuse_a=*/false, /*reuse_b=*/false);
    }

    // C/D layout: VGPR v -> row = v + 8*h, col = lane%16
#pragma unroll
    for (int v = 0; v < 8; ++v) {
        int row = ib + v + 8 * h;
        int col = jb + m;
        S[(size_t)row * TPAD + col] = acc[v] * INV_TEMP;
    }
}

// ---------------------------------------------------------------------------
// Kernel 4: per-row contrastive reduction (max -> neg_sum -> pos log-prob)
// ---------------------------------------------------------------------------
__global__ void ccel_rowstats(const float* __restrict__ S,
                              float* __restrict__ rowloss) {
    const int i = blockIdx.x;              // 0..987
    const int t = threadIdx.x;             // 0..127
    const int labi = (i / NVIEW) % NCLS;
    const float* row = S + (size_t)i * TPAD;
    __shared__ float red[128];

    // pass 1: row max (over all valid columns, like the reference)
    float m = -3.402823466e+38f;
    for (int j = t; j < TROWS; j += 128) m = fmaxf(m, row[j]);
    red[t] = m; __syncthreads();
    for (int off = 64; off >= 1; off >>= 1) {
        if (t < off) red[t] = fmaxf(red[t], red[t + off]);
        __syncthreads();
    }
    m = red[0]; __syncthreads();

    // pass 2: sum of exp(logit) over negatives (different class)
    float ns = 0.0f;
    for (int j = t; j < TROWS; j += 128) {
        int labj = (j / NVIEW) % NCLS;
        if (labj != labi) ns += __expf(row[j] - m);
    }
    red[t] = ns; __syncthreads();
    for (int off = 64; off >= 1; off >>= 1) {
        if (t < off) red[t] += red[t + off];
        __syncthreads();
    }
    ns = red[0]; __syncthreads();

    // pass 3: sum over positives (same class, j != i) of log-prob
    float ps = 0.0f;
    for (int j = t; j < TROWS; j += 128) {
        int labj = (j / NVIEW) % NCLS;
        if (labj == labi && j != i) {
            float lg = row[j] - m;
            ps += lg - __logf(__expf(lg) + ns);
        }
    }
    red[t] = ps; __syncthreads();
    for (int off = 64; off >= 1; off >>= 1) {
        if (t < off) red[t] += red[t + off];
        __syncthreads();
    }
    if (t == 0) {
        // positives per row: 2 batches * 26 views - self = 51
        float mlpp = red[0] / (float)(2 * NVIEW - 1);
        rowloss[i] = -LOSS_SCALE * mlpp;
    }
}

// ---------------------------------------------------------------------------
// Kernel 5: mean over rows -> scalar loss
// ---------------------------------------------------------------------------
__global__ void ccel_finalize(const float* __restrict__ rowloss,
                              float* __restrict__ out) {
    const int t = threadIdx.x;             // 0..255
    __shared__ float red[256];
    float s = 0.0f;
    for (int j = t; j < TROWS; j += 256) s += rowloss[j];
    red[t] = s; __syncthreads();
    for (int off = 128; off >= 1; off >>= 1) {
        if (t < off) red[t] += red[t + off];
        __syncthreads();
    }
    if (t == 0) out[0] = red[0] / (float)TROWS;
}

extern "C" void kernel_launch(void* const* d_in, const int* in_sizes, int n_in,
                              void* d_out, int out_size, void* d_ws, size_t ws_size,
                              hipStream_t stream) {
    const float* feats   = (const float*)d_in[0];
    const int*   labels  = (const int*)d_in[1];
    const int*   predict = (const int*)d_in[2];
    float*       out     = (float*)d_out;

    char* ws = (char*)d_ws;
    int*        idx     = (int*)ws;                                  // 38*26*4   = 3,952 B
    _Float16*   cf      = (_Float16*)(ws + 4096);                    // 992*256*2 = 507,904 B
    float*      S       = (float*)(ws + 4096 + 507904);              // 992*992*4 = 3,936,256 B
    float*      rowloss = (float*)(ws + 4096 + 507904 + 3936256);    // 988*4 B

    ccel_select  <<<NGROUP, 256, 0, stream>>>(labels, predict, idx);
    ccel_gather  <<<TPAD,    64, 0, stream>>>(feats, idx, cf);
    ccel_gemm    <<<dim3(TPAD / 16, TPAD / 16), 32, 0, stream>>>(cf, S);
    ccel_rowstats<<<TROWS,  128, 0, stream>>>(S, rowloss);
    ccel_finalize<<<1,      256, 0, stream>>>(rowloss, out);
}
